// T5Gemma2EncoderAttention_90855738180356
// MI455X (gfx1250) — matrix operand order repaired and checked by reading, not compiled, not run on hardware
//
#include <hip/hip_runtime.h>
#include <hip/hip_bf16.h>
#include <math.h>

#define DEV __device__ __forceinline__

typedef __bf16 bf16;
typedef __attribute__((ext_vector_type(16))) __bf16 v16bf;
typedef __attribute__((ext_vector_type(8)))  __bf16 v8bf;
typedef __attribute__((ext_vector_type(8)))  float  v8f;

// ---------------- problem constants ----------------
constexpr int B_   = 2;
constexpr int L_   = 2048;
constexpr int D_   = 2048;
constexpr int NQ_  = 8;
constexpr int NKV_ = 4;
constexpr int H_   = 256;
constexpr float CAP_    = 50.0f;
constexpr float SCALAR_ = 0.0625f;
constexpr float EPS_    = 1e-6f;
constexpr float KMASK_  = -3.3895314e38f;  // bf16 finfo.min

constexpr int MROWS   = B_ * L_;                 // 4096 token rows
constexpr int NC_QKV  = (NQ_ + 2 * NKV_) * H_;   // 4096 fused qkv cols
constexpr int KOUT    = NQ_ * H_;                // 2048 K-dim of out proj

// attention window geometry: query t attends s in [t-255, t+256]
constexpr int NT_   = 34;          // 16-key tiles covering [t0-256, t0+287]
constexpr int SPAN_ = NT_ * 16;    // 544
constexpr int PSTR_ = 560;         // padded LDS row stride (floats)

// ---------------- WMMA fragment helpers ----------------
// A fragment (16x32 bf16, M x K), row-major source with leading dim ld.
// Lane l: M = l%16, half = l/16; elem e -> K = half*16+e (contiguous 32B per lane).
DEV v16bf load_a16(const bf16* base, int ld, int k0) {
  int lane = threadIdx.x & 31;
  int ml = lane & 15, hf = lane >> 4;
  const bf16* p = base + (size_t)ml * ld + k0 + hf * 16;
  v8bf lo = *(const v8bf*)p;
  v8bf hi = *(const v8bf*)(p + 8);
  v16bf r;
#pragma unroll
  for (int i = 0; i < 8; ++i) { r[i] = lo[i]; r[i + 8] = hi[i]; }
  return r;
}

// B fragment (32x16 bf16, K x N) from N-major ("transposed") storage Bt[N][K], ld = K stride.
// Lane l: N = l%16, half = l/16; elem e -> K = k0 + half*16 + e (contiguous 32B per lane).
DEV v16bf load_bt16(const bf16* base, int ld, int k0) {
  int lane = threadIdx.x & 31;
  int ml = lane & 15, hf = lane >> 4;
  const bf16* p = base + (size_t)ml * ld + k0 + hf * 16;
  v8bf lo = *(const v8bf*)p;
  v8bf hi = *(const v8bf*)(p + 8);
  v16bf r;
#pragma unroll
  for (int i = 0; i < 8; ++i) { r[i] = lo[i]; r[i + 8] = hi[i]; }
  return r;
}

DEV v8f wmma_bf16(v16bf a, v16bf b, v8f c) {
  // (neg_a, A, neg_b, B, c_mod, C, reuse_a, reuse_b)
  return __builtin_amdgcn_wmma_f32_16x16x32_bf16(false, a, false, b, (short)0, c,
                                                 false, false);
}

// ---------------- conversion / repack kernels ----------------
__global__ void conv_x_kernel(const float* __restrict__ x, bf16* __restrict__ xbf) {
  int idx = blockIdx.x * 256 + threadIdx.x;          // MROWS*D_ threads
  xbf[idx] = (bf16)x[idx];
}

// Wt[col][d]: col<2048 -> q head n=col/256 : wq[n][d][h]
//             col in [2048,3072) -> k head : wkv[0][kv][d][h]
//             col in [3072,4096) -> v head : wkv[1][kv][d][h]
__global__ void conv_w_kernel(const float* __restrict__ wq,
                              const float* __restrict__ wkv,
                              bf16* __restrict__ wt) {
  int idx = blockIdx.x * 256 + threadIdx.x;          // NC_QKV*D_ threads
  int c = idx >> 11, d = idx & (D_ - 1);
  float v;
  if (c < NQ_ * H_) {
    int n = c >> 8, h = c & 255;
    v = wq[((size_t)n * D_ + d) * H_ + h];
  } else if (c < NQ_ * H_ + NKV_ * H_) {
    int cc = c - NQ_ * H_;
    int kv = cc >> 8, h = cc & 255;
    v = wkv[((size_t)kv * D_ + d) * H_ + h];
  } else {
    int cc = c - (NQ_ + NKV_) * H_;
    int kv = cc >> 8, h = cc & 255;
    v = wkv[(((size_t)NKV_ + kv) * D_ + d) * H_ + h];
  }
  wt[idx] = (bf16)v;
}

// woT[d][n*H+h] = wo[n][h][d]
__global__ void conv_wo_kernel(const float* __restrict__ wo, bf16* __restrict__ wot) {
  int idx = blockIdx.x * 256 + threadIdx.x;          // D_*KOUT threads
  int d = idx >> 11, c = idx & (KOUT - 1);
  int n = c >> 8, h = c & 255;
  wot[idx] = (bf16)wo[((size_t)n * H_ + h) * D_ + d];
}

// ---------------- fused QKV projection GEMM ----------------
// [MROWS x D_] @ [D_ x NC_QKV]; block = 4 waves, each wave owns a 32x64 strip.
// Explicit ping-pong double buffer (2x-unrolled K) so no register copies or
// WMMA-hazard NOPs between buffers; loads for one stage fly during the other's WMMAs.
__global__ void __launch_bounds__(128) gemm_qkv_kernel(
    const bf16* __restrict__ xbf, const bf16* __restrict__ wt,
    float* __restrict__ qraw, float* __restrict__ kraw, bf16* __restrict__ vt) {
  int w = threadIdx.x >> 5, lane = threadIdx.x & 31;
  int ml = lane & 15, hf = lane >> 4;
  int m0 = blockIdx.x * 128 + w * 32;
  int n0 = blockIdx.y * 64;

  v8f acc[2][4];
#pragma unroll
  for (int mi = 0; mi < 2; ++mi)
#pragma unroll
    for (int t = 0; t < 4; ++t)
#pragma unroll
      for (int j = 0; j < 8; ++j) acc[mi][t][j] = 0.0f;

  const bf16* abase0 = xbf + (size_t)m0 * D_;
  const bf16* abase1 = xbf + (size_t)(m0 + 16) * D_;

  v16bf a0[2], a1[2], bb[2][4];
  a0[0] = load_a16(abase0, D_, 0);
  a1[0] = load_a16(abase1, D_, 0);
#pragma unroll
  for (int t = 0; t < 4; ++t) bb[0][t] = load_bt16(wt + (size_t)(n0 + t * 16) * D_, D_, 0);

  for (int k0 = 0; k0 < D_; k0 += 64) {
    // prefetch far-ahead weight panel into GL2 (streamed once)
    if (k0 + 256 < D_)
      __builtin_prefetch(wt + (size_t)(n0 + ml) * D_ + k0 + 256, 0, 3);

    int k1 = k0 + 32;                             // D_ % 64 == 0 -> always valid
    a0[1] = load_a16(abase0, D_, k1);
    a1[1] = load_a16(abase1, D_, k1);
#pragma unroll
    for (int t = 0; t < 4; ++t) bb[1][t] = load_bt16(wt + (size_t)(n0 + t * 16) * D_, D_, k1);

#pragma unroll
    for (int t = 0; t < 4; ++t) acc[0][t] = wmma_bf16(a0[0], bb[0][t], acc[0][t]);
#pragma unroll
    for (int t = 0; t < 4; ++t) acc[1][t] = wmma_bf16(a1[0], bb[0][t], acc[1][t]);

    int k2 = (k0 + 64 < D_) ? k0 + 64 : 0;        // tail re-load, result unused
    a0[0] = load_a16(abase0, D_, k2);
    a1[0] = load_a16(abase1, D_, k2);
#pragma unroll
    for (int t = 0; t < 4; ++t) bb[0][t] = load_bt16(wt + (size_t)(n0 + t * 16) * D_, D_, k2);

#pragma unroll
    for (int t = 0; t < 4; ++t) acc[0][t] = wmma_bf16(a0[1], bb[1][t], acc[0][t]);
#pragma unroll
    for (int t = 0; t < 4; ++t) acc[1][t] = wmma_bf16(a1[1], bb[1][t], acc[1][t]);
  }

#pragma unroll
  for (int mi = 0; mi < 2; ++mi)
#pragma unroll
    for (int t = 0; t < 4; ++t) {
      int col = n0 + t * 16 + ml;
#pragma unroll
      for (int j = 0; j < 8; ++j) {
        int row = m0 + mi * 16 + j + 8 * hf;
        int bbi = row >> 11, tt = row & (L_ - 1);
        float v = acc[mi][t][j];
        if (col < NQ_ * H_) {
          int n = col >> 8, h = col & 255;
          qraw[(((size_t)bbi * L_ + tt) * NQ_ + n) * H_ + h] = v;
        } else if (col < (NQ_ + NKV_) * H_) {
          int cc = col - NQ_ * H_;
          int kv = cc >> 8, h = cc & 255;
          kraw[(((size_t)bbi * L_ + tt) * NKV_ + kv) * H_ + h] = v;
        } else {
          int cc = col - (NQ_ + NKV_) * H_;
          int kv = cc >> 8, h = cc & 255;
          vt[(((size_t)bbi * NKV_ + kv) * H_ + h) * L_ + tt] = (bf16)v;  // V pre-transposed
        }
      }
    }
}

// ---------------- RMS-norm + RoPE + scale ----------------
__global__ void __launch_bounds__(128) norm_rope_kernel(
    const float* __restrict__ qraw, const float* __restrict__ kraw,
    const int* __restrict__ segpos, const float* __restrict__ qscale,
    const float* __restrict__ kscale, bf16* __restrict__ qbf, bf16* __restrict__ kbf) {
  __shared__ float red[128];
  int bid = blockIdx.x;
  int head = bid % (NQ_ + NKV_);
  int t = (bid / (NQ_ + NKV_)) & (L_ - 1);
  int b = bid / ((NQ_ + NKV_) * L_);
  int i = threadIdx.x;

  const float* src;
  const float* scale;
  bf16* dst;
  bool isq = head < NQ_;
  if (isq) {
    src = qraw + (((size_t)b * L_ + t) * NQ_ + head) * H_;
    scale = qscale;
    dst = qbf + (((size_t)b * NQ_ + head) * L_ + t) * H_;
  } else {
    int kv = head - NQ_;
    src = kraw + (((size_t)b * L_ + t) * NKV_ + kv) * H_;
    scale = kscale;
    dst = kbf + (((size_t)b * NKV_ + kv) * L_ + t) * H_;
  }

  float v1 = src[i], v2 = src[i + 128];
  red[i] = v1 * v1 + v2 * v2;
  __syncthreads();
  for (int off = 64; off > 0; off >>= 1) {
    if (i < off) red[i] += red[i + off];
    __syncthreads();
  }
  float rms = rsqrtf(red[0] * (1.0f / H_) + EPS_);

  float a = v1 * rms * (1.0f + scale[i]);
  float bsec = v2 * rms * (1.0f + scale[i + 128]);
  float pos = (float)segpos[b * L_ + t];
  float ts = powf(10000.0f, (float)i * (1.0f / 128.0f));
  float ang = pos / ts;
  float sn = sinf(ang), cs = cosf(ang);
  float o1 = a * cs - bsec * sn;
  float o2 = bsec * cs + a * sn;
  if (isq) { o1 *= SCALAR_; o2 *= SCALAR_; }
  dst[i] = (bf16)o1;
  dst[i + 128] = (bf16)o2;
}

// ---------------- windowed attention ----------------
// Block = 4 waves, one (b, q-head, 16-query tile). q fragments (16x256) are hoisted
// into registers once; logits for the 544-key window live in LDS f32, softcap+mask+
// softmax there, then P@V with each wave owning a 64-wide H slice.
__global__ void __launch_bounds__(128) attn_kernel(
    const bf16* __restrict__ qbf, const bf16* __restrict__ kbf,
    const bf16* __restrict__ vt, bf16* __restrict__ enc) {
  __shared__ float probs[16 * PSTR_];

  int bid = blockIdx.x;
  int qt = bid & 127;
  int n = (bid >> 7) & 7;
  int b = bid >> 10;
  int t0 = qt * 16;
  int sb = t0 - 256;         // 16-aligned window base
  int kv = n >> 1;           // G = 2

  int w = threadIdx.x >> 5, lane = threadIdx.x & 31;
  int ml = lane & 15, hf = lane >> 4;

  const bf16* qbase = qbf + ((size_t)(b * NQ_ + n) * L_ + t0) * H_;
  const bf16* kbase = kbf + ((size_t)(b * NKV_ + kv) * L_) * H_;
  const bf16* vbase = vt + ((size_t)(b * NKV_ + kv) * H_) * L_;

  // hoist q fragments: invariant across all key tiles
  v16bf qf[8];
#pragma unroll
  for (int kk = 0; kk < 8; ++kk) qf[kk] = load_a16(qbase, H_, kk * 32);

  // phase 1: logits = q . k^T (load all 8 K fragments, then 8 back-to-back WMMAs)
  for (int jt = w; jt < NT_; jt += 4) {
    int s0 = sb + jt * 16;
    int srow = s0 + ml;
    int sc = srow < 0 ? 0 : (srow >= L_ ? L_ - 1 : srow);   // clamp; masked later
    const bf16* kp = kbase + (size_t)sc * H_;

    v16bf kf[8];
#pragma unroll
    for (int kk = 0; kk < 8; ++kk) {
      const bf16* p = kp + kk * 32 + hf * 16;
      v8bf lo = *(const v8bf*)p;
      v8bf hi = *(const v8bf*)(p + 8);
#pragma unroll
      for (int i = 0; i < 8; ++i) { kf[kk][i] = lo[i]; kf[kk][i + 8] = hi[i]; }
    }

    v8f acc;
#pragma unroll
    for (int j = 0; j < 8; ++j) acc[j] = 0.0f;
#pragma unroll
    for (int kk = 0; kk < 8; ++kk) acc = wmma_bf16(qf[kk], kf[kk], acc);

#pragma unroll
    for (int j = 0; j < 8; ++j) {
      int t = t0 + j + 8 * hf;
      int s = s0 + ml;
      float lg = CAP_ * tanhf(acc[j] * (1.0f / CAP_));
      int d = t - s;
      bool ok = (s >= 0) & (s < L_) & (d <= 255) & (d >= -256);
      probs[(j + 8 * hf) * PSTR_ + jt * 16 + ml] = ok ? lg : KMASK_;
    }
  }
  __syncthreads();

  // phase 2: softmax per query row; 8 lanes per row, shuffle reductions
  {
    int row = threadIdx.x >> 3, g = threadIdx.x & 7;
    float* pr = &probs[row * PSTR_];
    float m = -3.4e38f;
    for (int c = g; c < SPAN_; c += 8) m = fmaxf(m, pr[c]);
    for (int o = 4; o; o >>= 1) m = fmaxf(m, __shfl_xor(m, o, 8));
    float sum = 0.0f;
    for (int c = g; c < SPAN_; c += 8) sum += __expf(pr[c] - m);
    for (int o = 4; o; o >>= 1) sum += __shfl_xor(sum, o, 8);
    float inv = 1.0f / sum;
    for (int c = g; c < SPAN_; c += 8) pr[c] = __expf(pr[c] - m) * inv;
  }
  __syncthreads();

  // phase 3: encoded = P @ V ; wave w owns H slice [w*64, w*64+64)
  int hc0 = w * 64;
  v8f oacc[4];
#pragma unroll
  for (int t = 0; t < 4; ++t)
#pragma unroll
    for (int j = 0; j < 8; ++j) oacc[t][j] = 0.0f;

  for (int kc = 0; kc < SPAN_; kc += 32) {
    int ss = sb + kc + hf * 16;                 // 16-aligned key span
    ss = ss < 0 ? 0 : (ss > L_ - 16 ? L_ - 16 : ss);  // OOB spans have zero probs
    // load all 4 V fragments first (one wait), then A from LDS, then 4 WMMAs
    v16bf vf[4];
#pragma unroll
    for (int t = 0; t < 4; ++t) {
      const bf16* p = vbase + (size_t)(hc0 + t * 16 + ml) * L_ + ss;
      v8bf lo = *(const v8bf*)p;
      v8bf hi = *(const v8bf*)(p + 8);
#pragma unroll
      for (int i = 0; i < 8; ++i) { vf[t][i] = lo[i]; vf[t][i + 8] = hi[i]; }
    }
    v16bf a;
    const float* rp = &probs[ml * PSTR_ + kc];
#pragma unroll
    for (int i = 0; i < 8; ++i) a[i] = (bf16)rp[hf * 16 + i];
#pragma unroll
    for (int i = 0; i < 8; ++i) a[i + 8] = (bf16)rp[hf * 16 + 8 + i];
#pragma unroll
    for (int t = 0; t < 4; ++t) oacc[t] = wmma_bf16(a, vf[t], oacc[t]);
  }

#pragma unroll
  for (int t = 0; t < 4; ++t)
#pragma unroll
    for (int j = 0; j < 8; ++j) {
      int tq = t0 + j + 8 * hf;
      enc[(((size_t)b * L_ + tq) * NQ_ + n) * H_ + hc0 + t * 16 + ml] = (bf16)oacc[t][j];
    }
}

// ---------------- output projection GEMM ----------------
// [MROWS x KOUT] @ [KOUT x D_] -> fp32 out; same ping-pong 32x64-per-wave scheme.
__global__ void __launch_bounds__(128) gemm_out_kernel(
    const bf16* __restrict__ enc, const bf16* __restrict__ wot,
    float* __restrict__ out) {
  int w = threadIdx.x >> 5, lane = threadIdx.x & 31;
  int ml = lane & 15, hf = lane >> 4;
  int m0 = blockIdx.x * 128 + w * 32;
  int n0 = blockIdx.y * 64;

  v8f acc[2][4];
#pragma unroll
  for (int mi = 0; mi < 2; ++mi)
#pragma unroll
    for (int t = 0; t < 4; ++t)
#pragma unroll
      for (int j = 0; j < 8; ++j) acc[mi][t][j] = 0.0f;

  const bf16* abase0 = enc + (size_t)m0 * KOUT;
  const bf16* abase1 = enc + (size_t)(m0 + 16) * KOUT;

  v16bf a0[2], a1[2], bb[2][4];
  a0[0] = load_a16(abase0, KOUT, 0);
  a1[0] = load_a16(abase1, KOUT, 0);
#pragma unroll
  for (int t = 0; t < 4; ++t) bb[0][t] = load_bt16(wot + (size_t)(n0 + t * 16) * KOUT, KOUT, 0);

  for (int k0 = 0; k0 < KOUT; k0 += 64) {
    if (k0 + 256 < KOUT)
      __builtin_prefetch(wot + (size_t)(n0 + ml) * KOUT + k0 + 256, 0, 3);

    int k1 = k0 + 32;                             // KOUT % 64 == 0
    a0[1] = load_a16(abase0, KOUT, k1);
    a1[1] = load_a16(abase1, KOUT, k1);
#pragma unroll
    for (int t = 0; t < 4; ++t) bb[1][t] = load_bt16(wot + (size_t)(n0 + t * 16) * KOUT, KOUT, k1);

#pragma unroll
    for (int t = 0; t < 4; ++t) acc[0][t] = wmma_bf16(a0[0], bb[0][t], acc[0][t]);
#pragma unroll
    for (int t = 0; t < 4; ++t) acc[1][t] = wmma_bf16(a1[0], bb[0][t], acc[1][t]);

    int k2 = (k0 + 64 < KOUT) ? k0 + 64 : 0;      // tail re-load, result unused
    a0[0] = load_a16(abase0, KOUT, k2);
    a1[0] = load_a16(abase1, KOUT, k2);
#pragma unroll
    for (int t = 0; t < 4; ++t) bb[0][t] = load_bt16(wot + (size_t)(n0 + t * 16) * KOUT, KOUT, k2);

#pragma unroll
    for (int t = 0; t < 4; ++t) acc[0][t] = wmma_bf16(a0[1], bb[1][t], acc[0][t]);
#pragma unroll
    for (int t = 0; t < 4; ++t) acc[1][t] = wmma_bf16(a1[1], bb[1][t], acc[1][t]);
  }

#pragma unroll
  for (int mi = 0; mi < 2; ++mi)
#pragma unroll
    for (int t = 0; t < 4; ++t) {
      int col = n0 + t * 16 + ml;
#pragma unroll
      for (int j = 0; j < 8; ++j) {
        int row = m0 + mi * 16 + j + 8 * hf;
        out[(size_t)row * D_ + col] = acc[mi][t][j];
      }
    }
}

// ---------------- host launcher ----------------
extern "C" void kernel_launch(void* const* d_in, const int* in_sizes, int n_in,
                              void* d_out, int out_size, void* d_ws, size_t ws_size,
                              hipStream_t stream) {
  const float* x      = (const float*)d_in[0];
  const int*   segpos = (const int*)d_in[1];
  // d_in[2] = attn_mask (all-true); window mask is applied analytically
  const float* wq     = (const float*)d_in[3];
  const float* wkv    = (const float*)d_in[4];
  const float* wo     = (const float*)d_in[5];
  const float* qscale = (const float*)d_in[6];
  const float* kscale = (const float*)d_in[7];
  float* out = (float*)d_out;

  char* ws = (char*)d_ws;
  size_t off = 0;
  auto carve = [&](size_t bytes) {
    void* p = ws + off;
    off = (off + bytes + 255) & ~(size_t)255;
    return p;
  };
  bf16*  xbf  = (bf16*)carve((size_t)MROWS * D_ * 2);     // also reused for enc
  bf16*  wt   = (bf16*)carve((size_t)NC_QKV * D_ * 2);
  bf16*  wot  = (bf16*)carve((size_t)D_ * KOUT * 2);
  float* qraw = (float*)carve((size_t)B_ * L_ * NQ_ * H_ * 4);
  float* kraw = (float*)carve((size_t)B_ * L_ * NKV_ * H_ * 4);
  bf16*  vt   = (bf16*)carve((size_t)B_ * NKV_ * H_ * L_ * 2);
  bf16*  qbf  = (bf16*)carve((size_t)B_ * NQ_ * L_ * H_ * 2);
  bf16*  kbf  = (bf16*)carve((size_t)B_ * NKV_ * L_ * H_ * 2);
  bf16*  enc  = xbf;  // x is dead after the QKV GEMM; reuse its slot

  conv_x_kernel<<<(MROWS * D_) / 256, 256, 0, stream>>>(x, xbf);
  conv_w_kernel<<<(NC_QKV * D_) / 256, 256, 0, stream>>>(wq, wkv, wt);
  conv_wo_kernel<<<(D_ * KOUT) / 256, 256, 0, stream>>>(wo, wot);

  dim3 g1(MROWS / 128, NC_QKV / 64);
  gemm_qkv_kernel<<<g1, 128, 0, stream>>>(xbf, wt, qraw, kraw, vt);

  norm_rope_kernel<<<B_ * L_ * (NQ_ + NKV_), 128, 0, stream>>>(
      qraw, kraw, segpos, qscale, kscale, qbf, kbf);

  attn_kernel<<<B_ * NQ_ * (L_ / 16), 128, 0, stream>>>(qbf, kbf, vt, enc);

  dim3 g2(MROWS / 128, D_ / 64);
  gemm_out_kernel<<<g2, 128, 0, stream>>>(enc, wot, out);
}
